// SoftMoE_89094801588276
// MI455X (gfx1250) — matrix operand order repaired and checked by reading, not compiled, not run on hardware
//
#include <hip/hip_runtime.h>
#include <hip/hip_bf16.h>

// SoftMoE dims
#define BB 4
#define MM 1024
#define DD 768
#define EE 16
#define PP 64
#define FF 3072
#define NN (EE * PP)   // 1024 expert slots

typedef _Float16 f16;
typedef __attribute__((ext_vector_type(4)))  _Float16 v4h;
typedef __attribute__((ext_vector_type(16))) _Float16 v16h;
typedef __attribute__((ext_vector_type(8)))  float    v8f;

__device__ __forceinline__ float gelu_tanh(float x) {
    const float c = 0.7978845608028654f;  // sqrt(2/pi)
    float x3 = x * x * x;
    float t = tanhf(c * (x + 0.044715f * x3));
    return 0.5f * x * (1.0f + t);
}

// ---------------------------------------------------------------------------
// Tiled WMMA GEMM: C[z] = A[z] * B[z%bMod] (+ bias[z%bMod]) (opt GELU)
// f32 in memory, f16 in LDS, f32 accumulate (v_wmma_f32_16x16x32_f16).
// Register-staged, double-buffered LDS; 128-bit global loads.
// 256 threads = 8 wave32 waves (WROWS x WCOLS); each wave owns WM x WN frags.
// ---------------------------------------------------------------------------
template<int BM, int BN, int WROWS, int WCOLS, int WM, int WN, bool BIAS, bool GELU>
__global__ __launch_bounds__(256)
void softmoe_gemm_wmma(const float* __restrict__ A,  long long strideA, int lda,
                       const float* __restrict__ Bm, long long strideB, int bMod, int ldb,
                       const float* __restrict__ bias, long long strideBias,
                       float* __restrict__ C, long long strideC, int ldc,
                       int K)
{
    static_assert(WROWS * WM * 16 == BM, "BM mismatch");
    static_assert(WCOLS * WN * 16 == BN, "BN mismatch");
    static_assert(WROWS * WCOLS == 8, "8 waves per block");
    constexpr int BK   = 32;
    constexpr int PAD  = 8;
    constexpr int AQ   = BK / 4;            // float4 quads per A row
    constexpr int NQ   = BN / 4;            // float4 quads per B row
    constexpr int AL4  = BM * BK / (256 * 4);
    constexpr int BL4  = BN * BK / (256 * 4);

    __shared__ f16 As [2][BM][BK + PAD];    // (m, k) row-major
    __shared__ f16 BsT[2][BN][BK + PAD];    // (n, k) k-transposed

    const int z  = blockIdx.z;
    const float* Ab = A  + (long long)z * strideA;
    const float* Bb = Bm + (long long)(z % bMod) * strideB;
    float*       Cb = C  + (long long)z * strideC;
    const float* biasp = nullptr;
    if constexpr (BIAS) biasp = bias + (long long)(z % bMod) * strideBias;

    const int m0 = blockIdx.y * BM;
    const int n0 = blockIdx.x * BN;

    const int tid  = threadIdx.x;
    const int lane = tid & 31;
    const int wid  = tid >> 5;
    const int lr   = lane & 15;             // frag row (A) / frag col (B, C)
    const int hg   = lane >> 4;             // half-group

    const int wm0 = (wid % WROWS) * (WM * 16);
    const int wn0 = (wid / WROWS) * (WN * 16);

    // per-thread staging coordinates (compile-time unrolled)
    int aR[AL4], aQ[AL4], bR[BL4], bQ[BL4];
    const float* aSrc[AL4];
    const float* bSrc[BL4];
    #pragma unroll
    for (int i = 0; i < AL4; i++) {
        int idx = tid + i * 256;
        aR[i] = idx / AQ; aQ[i] = idx % AQ;
        aSrc[i] = Ab + (long long)(m0 + aR[i]) * lda + aQ[i] * 4;
    }
    #pragma unroll
    for (int i = 0; i < BL4; i++) {
        int idx = tid + i * 256;
        bR[i] = idx / NQ; bQ[i] = idx % NQ;   // bR = k, bQ = n quad
        bSrc[i] = Bb + (long long)bR[i] * ldb + n0 + bQ[i] * 4;
    }

    v8f acc[WM][WN];
    #pragma unroll
    for (int i = 0; i < WM; i++)
        #pragma unroll
        for (int j = 0; j < WN; j++)
            #pragma unroll
            for (int v = 0; v < 8; v++) acc[i][j][v] = 0.0f;

    float4 aReg[AL4], bReg[BL4];

    // ---- staging helpers (macros keep everything in registers) ----
#define LOAD_G(KT)                                                          \
    do {                                                                    \
        _Pragma("unroll")                                                   \
        for (int i = 0; i < AL4; i++) aReg[i] = *(const float4*)(aSrc[i] + (KT)); \
        _Pragma("unroll")                                                   \
        for (int i = 0; i < BL4; i++) bReg[i] = *(const float4*)(bSrc[i] + (long long)(KT) * ldb); \
    } while (0)

#define STORE_L(BUF)                                                        \
    do {                                                                    \
        _Pragma("unroll")                                                   \
        for (int i = 0; i < AL4; i++) {                                     \
            v4h h; h[0] = (f16)aReg[i].x; h[1] = (f16)aReg[i].y;            \
                   h[2] = (f16)aReg[i].z; h[3] = (f16)aReg[i].w;            \
            *(v4h*)&As[BUF][aR[i]][aQ[i] * 4] = h;                          \
        }                                                                   \
        _Pragma("unroll")                                                   \
        for (int i = 0; i < BL4; i++) {                                     \
            BsT[BUF][bQ[i] * 4 + 0][bR[i]] = (f16)bReg[i].x;                \
            BsT[BUF][bQ[i] * 4 + 1][bR[i]] = (f16)bReg[i].y;                \
            BsT[BUF][bQ[i] * 4 + 2][bR[i]] = (f16)bReg[i].z;                \
            BsT[BUF][bQ[i] * 4 + 3][bR[i]] = (f16)bReg[i].w;                \
        }                                                                   \
    } while (0)

    const int nTiles = K / BK;

    // prologue: tile 0
    LOAD_G(0);
    STORE_L(0);
    __syncthreads();

    for (int t = 0; t < nTiles; t++) {
        const int cur = t & 1;
        const int kn  = (t + 1) * BK;

        if (t + 1 < nTiles) {
            LOAD_G(kn);                                     // in-flight during WMMA
            if (t + 2 < nTiles) {                           // warm L2 two tiles ahead
                __builtin_prefetch((const void*)(aSrc[0] + kn + BK), 0, 2);
                __builtin_prefetch((const void*)(bSrc[0] + (long long)(kn + BK) * ldb), 0, 2);
            }
        }

        // A fragments: m = lr; e<8 -> k = hg*8+e ; e>=8 -> k = 16+hg*8+(e-8)
        v16h afr[WM];
        #pragma unroll
        for (int i = 0; i < WM; i++) {
            const f16* ar = &As[cur][wm0 + i * 16 + lr][0];
            #pragma unroll
            for (int e = 0; e < 8; e++) afr[i][e]     = ar[hg * 8 + e];
            #pragma unroll
            for (int e = 0; e < 8; e++) afr[i][8 + e] = ar[16 + hg * 8 + e];
        }
        // B fragments: n = lr; e<8 -> k = hg*8+e ; e>=8 -> k = 16+hg*8+(e-8)
        v16h bfr[WN];
        #pragma unroll
        for (int j = 0; j < WN; j++) {
            const f16* br = &BsT[cur][wn0 + j * 16 + lr][0];
            #pragma unroll
            for (int e = 0; e < 8; e++) bfr[j][e]     = br[hg * 8 + e];
            #pragma unroll
            for (int e = 0; e < 8; e++) bfr[j][8 + e] = br[16 + hg * 8 + e];
        }

        #pragma unroll
        for (int i = 0; i < WM; i++)
            #pragma unroll
            for (int j = 0; j < WN; j++)
                acc[i][j] = __builtin_amdgcn_wmma_f32_16x16x32_f16(
                    false, afr[i], false, bfr[j], (short)0, acc[i][j], false, false);

        if (t + 1 < nTiles) STORE_L(1 - cur);
        __syncthreads();
    }
#undef LOAD_G
#undef STORE_L

    // epilogue: C/D layout — n = lr across lanes, m = hg*8 + v per VGPR
    #pragma unroll
    for (int i = 0; i < WM; i++) {
        #pragma unroll
        for (int j = 0; j < WN; j++) {
            const int n = n0 + wn0 + j * 16 + lr;
            float bv = 0.0f;
            if constexpr (BIAS) bv = biasp[n];
            #pragma unroll
            for (int v = 0; v < 8; v++) {
                const int m = m0 + wm0 + i * 16 + hg * 8 + v;
                float val = acc[i][j][v] + bv;
                if constexpr (GELU) val = gelu_tanh(val);
                Cb[(long long)m * ldc + n] = val;
            }
        }
    }
}

// ---------------------------------------------------------------------------
// dispatch = softmax over tokens (axis m) for each (b, n) column of logits.
// Writes the result TRANSPOSED: dispT[b][n][m]  (free transpose for mix GEMM)
// ---------------------------------------------------------------------------
__global__ __launch_bounds__(256)
void softmoe_dispatch_softmax(const float* __restrict__ logits, float* __restrict__ dispT)
{
    const int bn = blockIdx.x;            // b*NN + n
    const int b  = bn >> 10;
    const int n  = bn & (NN - 1);
    const int t  = threadIdx.x;
    const float* col = logits + ((long long)b * MM * NN) + n;

    __shared__ float red[256];
    float v[4];
    float mx = -3.4e38f;
    #pragma unroll
    for (int i = 0; i < 4; i++) { v[i] = col[(long long)(t + i * 256) * NN]; mx = fmaxf(mx, v[i]); }
    red[t] = mx; __syncthreads();
    for (int s = 128; s > 0; s >>= 1) { if (t < s) red[t] = fmaxf(red[t], red[t + s]); __syncthreads(); }
    mx = red[0]; __syncthreads();

    float sum = 0.0f;
    #pragma unroll
    for (int i = 0; i < 4; i++) { v[i] = __expf(v[i] - mx); sum += v[i]; }
    red[t] = sum; __syncthreads();
    for (int s = 128; s > 0; s >>= 1) { if (t < s) red[t] += red[t + s]; __syncthreads(); }
    const float inv = 1.0f / red[0];

    float* orow = dispT + (long long)bn * MM;
    #pragma unroll
    for (int i = 0; i < 4; i++) orow[t + i * 256] = v[i] * inv;
}

// ---------------------------------------------------------------------------
// combine = softmax over (e,p) for each (b, m) row of logits; also emits
// probabilities[b][m][e] = mean_p combine  and  top_experts = argmax_e.
// ---------------------------------------------------------------------------
__global__ __launch_bounds__(256)
void softmoe_combine_softmax(const float* __restrict__ logits, float* __restrict__ combine,
                             float* __restrict__ probs, float* __restrict__ tope)
{
    const int bm = blockIdx.x;            // b*MM + m
    const int t  = threadIdx.x;
    const float* row = logits + (long long)bm * NN;

    __shared__ float ex[NN];
    __shared__ float red[256];

    float v[4];
    float mx = -3.4e38f;
    #pragma unroll
    for (int i = 0; i < 4; i++) { v[i] = row[t + i * 256]; mx = fmaxf(mx, v[i]); }
    red[t] = mx; __syncthreads();
    for (int s = 128; s > 0; s >>= 1) { if (t < s) red[t] = fmaxf(red[t], red[t + s]); __syncthreads(); }
    mx = red[0]; __syncthreads();

    float sum = 0.0f;
    #pragma unroll
    for (int i = 0; i < 4; i++) { v[i] = __expf(v[i] - mx); sum += v[i]; ex[t + i * 256] = v[i]; }
    red[t] = sum; __syncthreads();
    for (int s = 128; s > 0; s >>= 1) { if (t < s) red[t] += red[t + s]; __syncthreads(); }
    const float inv = 1.0f / red[0];

    #pragma unroll
    for (int i = 0; i < 4; i++) combine[(long long)bm * NN + t + i * 256] = v[i] * inv;
    __syncthreads();

    if (t < EE) {
        float s = 0.0f;
        #pragma unroll
        for (int p = 0; p < PP; p++) s += ex[t * PP + p];
        const float pr = s * inv * (1.0f / (float)PP);
        red[t] = pr;
        probs[(long long)bm * EE + t] = pr;
    }
    __syncthreads();
    if (t == 0) {
        int best = 0; float bv = red[0];
        for (int e = 1; e < EE; e++) if (red[e] > bv) { bv = red[e]; best = e; }
        tope[bm] = (float)best;
    }
}

// ---------------------------------------------------------------------------
extern "C" void kernel_launch(void* const* d_in, const int* in_sizes, int n_in,
                              void* d_out, int out_size, void* d_ws, size_t ws_size,
                              hipStream_t stream)
{
    const float* x   = (const float*)d_in[0];   // [B,M,D]
    const float* phi = (const float*)d_in[1];   // [D,E,P] = [D,N]
    const float* W1  = (const float*)d_in[2];   // [E,D,F]
    const float* b1  = (const float*)d_in[3];   // [E,F]
    const float* W2  = (const float*)d_in[4];   // [E,F,D]
    const float* b2  = (const float*)d_in[5];   // [E,D]

    float* out = (float*)d_out;
    float* outputs = out;                                        // [B,M,D]
    float* probs   = outputs + (long long)BB * MM * DD;          // [B,M,E]
    float* tope    = probs   + (long long)BB * MM * EE;          // [B,M,1]
    float* hidden  = tope    + (long long)BB * MM;               // [B,N,F]

    float* ws      = (float*)d_ws;
    float* logits  = ws;                                         // [B,M,N]
    float* dispT   = logits  + (long long)BB * MM * NN;          // [B,N,M]
    float* combine = dispT   + (long long)BB * NN * MM;          // [B,M,N]
    float* mix     = combine + (long long)BB * MM * NN;          // [B,N,D]
    float* eout    = mix     + (long long)BB * NN * DD;          // [B,N,D]

    const dim3 blk(256);

    // 1) logits[B*M, N] = x[B*M, D] * phi[D, N]
    softmoe_gemm_wmma<128, 128, 4, 2, 2, 4, false, false>
        <<<dim3(NN / 128, (BB * MM) / 128, 1), blk, 0, stream>>>(
            x, 0, DD, phi, 0, 1, NN, nullptr, 0, logits, 0, NN, DD);

    // 2) dispatch = softmax over tokens (stored transposed)
    softmoe_dispatch_softmax<<<dim3(BB * NN), blk, 0, stream>>>(logits, dispT);

    // 3) combine = softmax over slots; + probabilities + top_experts
    softmoe_combine_softmax<<<dim3(BB * MM), blk, 0, stream>>>(logits, combine, probs, tope);

    // 4) mix[b][N, D] = dispT[b][N, M] * x[b][M, D]
    softmoe_gemm_wmma<128, 128, 4, 2, 2, 4, false, false>
        <<<dim3(DD / 128, NN / 128, BB), blk, 0, stream>>>(
            dispT, (long long)NN * MM, MM,
            x,     (long long)MM * DD, BB, DD,
            nullptr, 0,
            mix,   (long long)NN * DD, DD, MM);

    // 5) hidden[z=(b,e)][64, F] = gelu(mix_slice[64, D] * W1[e][D, F] + b1[e])
    softmoe_gemm_wmma<64, 128, 2, 4, 2, 2, true, true>
        <<<dim3(FF / 128, 1, BB * EE), blk, 0, stream>>>(
            mix, (long long)PP * DD, DD,
            W1,  (long long)DD * FF, EE, FF,
            b1,  (long long)FF,
            hidden, (long long)PP * FF, FF, DD);

    // 6) eout[z=(b,e)][64, D] = hidden_slice[64, F] * W2[e][F, D] + b2[e]
    softmoe_gemm_wmma<64, 128, 2, 4, 2, 2, true, false>
        <<<dim3(DD / 128, 1, BB * EE), blk, 0, stream>>>(
            hidden, (long long)PP * FF, FF,
            W2,     (long long)FF * DD, EE, DD,
            b2,     (long long)DD,
            eout,   (long long)PP * DD, DD, FF);

    // 7) outputs[b][M, D] = combine[b][M, N] * eout[b][N, D]
    softmoe_gemm_wmma<128, 128, 4, 2, 2, 4, false, false>
        <<<dim3(DD / 128, MM / 128, BB), blk, 0, stream>>>(
            combine, (long long)MM * NN, NN,
            eout,    (long long)NN * DD, BB, DD,
            nullptr, 0,
            outputs, (long long)MM * DD, DD, NN);
}
